// GCNWithTransformer_91010357002765
// MI455X (gfx1250) — compile-verified, matrix operand order checked by reading, not run on hardware
//
#include <hip/hip_runtime.h>
#include <hip/hip_bf16.h>

typedef __attribute__((ext_vector_type(16))) __bf16 v16bf;
typedef __attribute__((ext_vector_type(8)))  float  v8f;

#define TPB 256

// ---------------------------------------------------------------------------
// Elementwise helpers
// ---------------------------------------------------------------------------
__global__ void fill_f32(float* __restrict__ p, float v, size_t n) {
  size_t i = (size_t)blockIdx.x * blockDim.x + threadIdx.x;
  if (i < n) p[i] = v;
}

__global__ void cvt_f32_bf16(const float* __restrict__ in, __bf16* __restrict__ out, size_t n) {
  size_t i = (size_t)blockIdx.x * blockDim.x + threadIdx.x;
  if (i < n) out[i] = (__bf16)in[i];
}

// ---------------------------------------------------------------------------
// Degree / normalization for GCN
// ---------------------------------------------------------------------------
__global__ void deg_edges(const int* __restrict__ dst, const float* __restrict__ ew,
                          float* __restrict__ deg, int E) {
  int e = blockIdx.x * blockDim.x + threadIdx.x;
  if (e < E) unsafeAtomicAdd(&deg[dst[e]], ew[e]);
}

__global__ void make_dis(const float* __restrict__ deg, float* __restrict__ dis, int Nn) {
  int i = blockIdx.x * blockDim.x + threadIdx.x;
  if (i < Nn) {
    float d = deg[i];
    dis[i] = d > 0.0f ? rsqrtf(d) : 0.0f;
  }
}

// per-edge normalization coefficient, computed once (removes 3 dependent
// gathers from the hot scatter-add kernel)
__global__ void edge_norms(const int* __restrict__ src, const int* __restrict__ dst,
                           const float* __restrict__ ew, const float* __restrict__ dis,
                           float* __restrict__ enorm, int E) {
  int e = blockIdx.x * blockDim.x + threadIdx.x;
  if (e < E) enorm[e] = dis[src[e]] * ew[e] * dis[dst[e]];
}

// ---------------------------------------------------------------------------
// WMMA bf16 GEMM:  C[M,Nc] = A[M,K] @ B[Nc,K]^T   (NT, both K-contiguous)
// block = 256 threads = 8 waves arranged 4 (rows) x 2 (cols);
// wave tile = 32 x 64  ->  block tile = 128 x 128.
// Per 32-wide k-step: 2 A-frag + 4 B-frag loads feed 8 WMMAs.
// ---------------------------------------------------------------------------
__global__ __launch_bounds__(256)
void wmma_gemm_bf16_nt(const __bf16* __restrict__ A, const __bf16* __restrict__ B,
                       float* __restrict__ C, int M, int K, int Nc) {
  const int lane  = threadIdx.x & 31;
  const int wave  = threadIdx.x >> 5;
  const int rg    = wave >> 1;      // row group 0..3
  const int cg    = wave & 1;       // col group 0..1
  const int m16   = lane & 15;      // M row (A) / N col (B) within 16
  const int khalf = lane >> 4;      // which 16-wide K half this lane owns

  const int row_base = blockIdx.x * 128 + rg * 32;
  const int col_base = blockIdx.y * 128 + cg * 64;

  int ar0 = row_base + m16;      if (ar0 >= M) ar0 = M - 1;   // clamp tail
  int ar1 = row_base + 16 + m16; if (ar1 >= M) ar1 = M - 1;   // (stores guarded)
  const __bf16* ap0 = A + (size_t)ar0 * K + khalf * 16;
  const __bf16* ap1 = A + (size_t)ar1 * K + khalf * 16;

  const __bf16* bp[4];
#pragma unroll
  for (int t = 0; t < 4; ++t)
    bp[t] = B + (size_t)(col_base + t * 16 + m16) * K + khalf * 16;

  v8f acc[2][4] = {};

  for (int k = 0; k < K; k += 32) {
    __builtin_prefetch(ap0 + k + 256, 0, 1);      // global_prefetch_b8, stream A
    v16bf a0 = *(const v16bf*)(ap0 + k);
    v16bf a1 = *(const v16bf*)(ap1 + k);
#pragma unroll
    for (int t = 0; t < 4; ++t) {
      v16bf b = *(const v16bf*)(bp[t] + k);
      acc[0][t] = __builtin_amdgcn_wmma_f32_16x16x32_bf16(
          false, a0, false, b, (short)0, acc[0][t], false, false);
      acc[1][t] = __builtin_amdgcn_wmma_f32_16x16x32_bf16(
          false, a1, false, b, (short)0, acc[1][t], false, false);
    }
  }

#pragma unroll
  for (int i = 0; i < 2; ++i) {
#pragma unroll
    for (int t = 0; t < 4; ++t) {
      int col = col_base + t * 16 + m16;
#pragma unroll
      for (int v = 0; v < 8; ++v) {
        int row = row_base + i * 16 + v + khalf * 8;  // C layout: VGPR v -> M = v + 8*khalf
        if (row < M && col < Nc) C[(size_t)row * Nc + col] = acc[i][t][v];
      }
    }
  }
}

// ---------------------------------------------------------------------------
// GCN edge scatter-add:  out[dst] += xt[src] * enorm[e];  grid = (E, Hf/256)
// ---------------------------------------------------------------------------
__global__ void edge_aggregate(const float* __restrict__ xt, float* __restrict__ out,
                               const int* __restrict__ src, const int* __restrict__ dst,
                               const float* __restrict__ enorm, int Hf) {
  int e = blockIdx.x;
  int f = blockIdx.y * blockDim.x + threadIdx.x;
  int s = src[e], d = dst[e];
  float v = xt[(size_t)s * Hf + f] * enorm[e];
  unsafeAtomicAdd(&out[(size_t)d * Hf + f], v);
}

// self-loop term + bias + relu:  out = relu(out + xt*dis^2 + bias)
__global__ void agg_finish(float* __restrict__ out, const float* __restrict__ xt,
                           const float* __restrict__ dis, const float* __restrict__ bias,
                           int Nn, int Hf) {
  size_t i = (size_t)blockIdx.x * blockDim.x + threadIdx.x;
  if (i >= (size_t)Nn * Hf) return;
  int r = (int)(i / Hf), c = (int)(i % Hf);
  float d = dis[r];
  float v = out[i] + xt[i] * d * d + bias[c];
  out[i] = v > 0.0f ? v : 0.0f;
}

// ---------------------------------------------------------------------------
// BatchNorm over columns (Hf == 1024): per-block partial sums, f32 atomics
// ---------------------------------------------------------------------------
__global__ __launch_bounds__(256)
void bn_colstats(const float* __restrict__ x, float* __restrict__ ssum,
                 float* __restrict__ ssq, int Nn, int rows_per_block) {
  const int Hf = 1024;
  float s[4] = {0, 0, 0, 0}, q[4] = {0, 0, 0, 0};
  int r0 = blockIdx.x * rows_per_block;
  int r1 = r0 + rows_per_block; if (r1 > Nn) r1 = Nn;
  for (int r = r0; r < r1; ++r) {
    const float* row = x + (size_t)r * Hf;
#pragma unroll
    for (int j = 0; j < 4; ++j) {
      float v = row[threadIdx.x + j * 256];
      s[j] += v; q[j] += v * v;
    }
  }
#pragma unroll
  for (int j = 0; j < 4; ++j) {
    unsafeAtomicAdd(&ssum[threadIdx.x + j * 256], s[j]);
    unsafeAtomicAdd(&ssq [threadIdx.x + j * 256], q[j]);
  }
}

__global__ void bn_normalize_bf16(const float* __restrict__ x, const float* __restrict__ ssum,
                                  const float* __restrict__ ssq, const float* __restrict__ g,
                                  const float* __restrict__ b, __bf16* __restrict__ out,
                                  int Nn, int Hf) {
  size_t i = (size_t)blockIdx.x * blockDim.x + threadIdx.x;
  if (i >= (size_t)Nn * Hf) return;
  int c = (int)(i % Hf);
  float m   = ssum[c] / (float)Nn;
  float var = ssq[c] / (float)Nn - m * m;
  float inv = rsqrtf(var + 1e-5f);
  out[i] = (__bf16)((x[i] - m) * inv * g[c] + b[c]);
}

// ---------------------------------------------------------------------------
// GEMM epilogues
// ---------------------------------------------------------------------------
__global__ void epilogue_bf16(const float* __restrict__ G, const float* __restrict__ bias,
                              __bf16* __restrict__ out, int Nn, int Nc, int relu) {
  size_t i = (size_t)blockIdx.x * blockDim.x + threadIdx.x;
  if (i >= (size_t)Nn * Nc) return;
  int c = (int)(i % Nc);
  float v = G[i] + bias[c];
  if (relu && v < 0.0f) v = 0.0f;
  out[i] = (__bf16)v;
}

__global__ void epilogue_f32_bf16(const float* __restrict__ G, const float* __restrict__ bias,
                                  float* __restrict__ Hout, __bf16* __restrict__ XBout,
                                  int Nn, int Nc) {
  size_t i = (size_t)blockIdx.x * blockDim.x + threadIdx.x;
  if (i >= (size_t)Nn * Nc) return;
  int c = (int)(i % Nc);
  float v = G[i] + bias[c];
  Hout[i]  = v;
  XBout[i] = (__bf16)v;
}

// ---------------------------------------------------------------------------
// Fused residual + bias + LayerNorm (D=512), one wave32 per row
// H = LN(H + G + badd) * gamma + beta ; also emit bf16 copy for next GEMM
// ---------------------------------------------------------------------------
__global__ __launch_bounds__(256)
void ln_fused(float* __restrict__ Hio, const float* __restrict__ G,
              const float* __restrict__ badd, const float* __restrict__ gamma,
              const float* __restrict__ beta, __bf16* __restrict__ XBout, int Nn) {
  const int D = 512;
  int lane = threadIdx.x & 31;
  int wave = threadIdx.x >> 5;
  int r = blockIdx.x * 8 + wave;
  if (r >= Nn) return;

  float vals[16];
  float s = 0.0f, q = 0.0f;
  const float* hrow = Hio + (size_t)r * D;
  const float* grow = G   + (size_t)r * D;
#pragma unroll
  for (int j = 0; j < 16; ++j) {
    int c = j * 32 + lane;
    float v = hrow[c] + grow[c] + badd[c];
    vals[j] = v; s += v; q += v * v;
  }
#pragma unroll
  for (int off = 16; off > 0; off >>= 1) {
    s += __shfl_xor(s, off, 32);
    q += __shfl_xor(q, off, 32);
  }
  float m   = s / (float)D;
  float inv = rsqrtf(q / (float)D - m * m + 1e-5f);
  float*  hw = Hio   + (size_t)r * D;
  __bf16* xw = XBout + (size_t)r * D;
#pragma unroll
  for (int j = 0; j < 16; ++j) {
    int c = j * 32 + lane;
    float o = (vals[j] - m) * inv * gamma[c] + beta[c];
    hw[c] = o; xw[c] = (__bf16)o;
  }
}

// ---------------------------------------------------------------------------
// Final tiny GEMM (Nc = 10): one thread per row
// ---------------------------------------------------------------------------
__global__ void fc1_kernel(const __bf16* __restrict__ XB, const float* __restrict__ W,
                           const float* __restrict__ b, float* __restrict__ out, int Nn) {
  int r = blockIdx.x * blockDim.x + threadIdx.x;
  if (r >= Nn) return;
  const __bf16* row = XB + (size_t)r * 512;
  float acc[10];
#pragma unroll
  for (int o = 0; o < 10; ++o) acc[o] = b[o];
  for (int k = 0; k < 512; ++k) {
    float a = (float)row[k];
#pragma unroll
    for (int o = 0; o < 10; ++o) acc[o] += a * W[o * 512 + k];
  }
#pragma unroll
  for (int o = 0; o < 10; ++o) out[(size_t)r * 10 + o] = acc[o];
}

// ---------------------------------------------------------------------------
// Host-side orchestration
// ---------------------------------------------------------------------------
static inline size_t al256(size_t x) { return (x + 255) & ~(size_t)255; }
static inline int gblk(size_t n) { return (int)((n + TPB - 1) / TPB); }

extern "C" void kernel_launch(void* const* d_in, const int* in_sizes, int n_in,
                              void* d_out, int out_size, void* d_ws, size_t ws_size,
                              hipStream_t stream) {
  const int Nn = 20000, Ecnt = 320000;
  const int D = 512, Hh = 1024, DFF = 2048, Lc = 2;

  // --- inputs (setup_inputs order) ---
  const float* x        = (const float*)d_in[0];
  const int*   ei       = (const int*)  d_in[1];
  const float* ew       = (const float*)d_in[2];
  const float* gcn1_w   = (const float*)d_in[3];
  const float* gcn1_b   = (const float*)d_in[4];
  const float* bn1_g    = (const float*)d_in[5];
  const float* bn1_b    = (const float*)d_in[6];
  const float* gcn2_w   = (const float*)d_in[7];
  const float* gcn2_b   = (const float*)d_in[8];
  const float* bn2_g    = (const float*)d_in[9];
  const float* bn2_b    = (const float*)d_in[10];
  const float* proj_w   = (const float*)d_in[11];
  const float* proj_b   = (const float*)d_in[12];
  const float* t_qkv_w  = (const float*)d_in[13];
  const float* t_qkv_b  = (const float*)d_in[14];
  const float* t_out_w  = (const float*)d_in[15];
  const float* t_out_b  = (const float*)d_in[16];
  const float* t_ln1_g  = (const float*)d_in[17];
  const float* t_ln1_b  = (const float*)d_in[18];
  const float* t_ff1_w  = (const float*)d_in[19];
  const float* t_ff1_b  = (const float*)d_in[20];
  const float* t_ff2_w  = (const float*)d_in[21];
  const float* t_ff2_b  = (const float*)d_in[22];
  const float* t_ln2_g  = (const float*)d_in[23];
  const float* t_ln2_b  = (const float*)d_in[24];
  const float* fc_w     = (const float*)d_in[25];
  const float* fc_b     = (const float*)d_in[26];
  const float* fc1_w    = (const float*)d_in[27];
  const float* fc1_b    = (const float*)d_in[28];
  (void)in_sizes; (void)n_in; (void)out_size; (void)ws_size;

  const int* esrc = ei;
  const int* edst = ei + Ecnt;

  // --- workspace carve-up ---
  char* ws = (char*)d_ws;
  size_t off = 0;
  auto carve = [&](size_t bytes) { void* p = ws + off; off = al256(off + bytes); return p; };
  float*  G     = (float*) carve((size_t)Nn * 2048 * 4);  // GEMM output (max N x 2048)
  float*  T     = (float*) carve((size_t)Nn * 1024 * 4);  // aggregation temp
  float*  Hb    = (float*) carve((size_t)Nn * 512  * 4);  // persistent h (f32)
  __bf16* XB    = (__bf16*)carve((size_t)Nn * 2048 * 2);  // bf16 activations
  float*  deg   = (float*) carve((size_t)Nn * 4);
  float*  dis   = (float*) carve((size_t)Nn * 4);
  float*  enorm = (float*) carve((size_t)Ecnt * 4);
  float*  ssum  = (float*) carve(1024 * 4);
  float*  ssq   = (float*) carve(1024 * 4);

  auto wcarve = [&](size_t elems) { return (__bf16*)carve(elems * 2); };
  __bf16* wg1   = wcarve((size_t)Hh * D);
  __bf16* wg2   = wcarve((size_t)Hh * Hh);
  __bf16* wproj = wcarve((size_t)D * Hh);
  __bf16* wv[2], *wout[2], *wff1[2], *wff2[2];
  for (int l = 0; l < Lc; ++l) {
    wv[l]   = wcarve((size_t)D * D);
    wout[l] = wcarve((size_t)D * D);
    wff1[l] = wcarve((size_t)DFF * D);
    wff2[l] = wcarve((size_t)D * DFF);
  }
  __bf16* wfc = wcarve((size_t)D * D);

  auto cvt = [&](const float* src, __bf16* dst, size_t n) {
    cvt_f32_bf16<<<gblk(n), TPB, 0, stream>>>(src, dst, n);
  };
  auto gemm = [&](const __bf16* A, const __bf16* B, float* C, int M, int K, int Nc) {
    dim3 grid((M + 127) / 128, Nc / 128);
    wmma_gemm_bf16_nt<<<grid, 256, 0, stream>>>(A, B, C, M, K, Nc);
  };

  // --- weight conversion to bf16 (small, L2-resident) ---
  cvt(gcn1_w, wg1,   (size_t)Hh * D);
  cvt(gcn2_w, wg2,   (size_t)Hh * Hh);
  cvt(proj_w, wproj, (size_t)D * Hh);
  for (int l = 0; l < Lc; ++l) {
    cvt(t_qkv_w + (size_t)l * 3 * D * D + (size_t)2 * D * D, wv[l], (size_t)D * D);
    cvt(t_out_w + (size_t)l * D * D,    wout[l], (size_t)D * D);
    cvt(t_ff1_w + (size_t)l * DFF * D,  wff1[l], (size_t)DFF * D);
    cvt(t_ff2_w + (size_t)l * D * DFF,  wff2[l], (size_t)D * DFF);
  }
  cvt(fc_w, wfc, (size_t)D * D);

  // --- GCN degree normalization ---
  fill_f32<<<gblk(Nn), TPB, 0, stream>>>(deg, 1.0f, (size_t)Nn);      // self loops
  deg_edges<<<gblk(Ecnt), TPB, 0, stream>>>(edst, ew, deg, Ecnt);
  make_dis<<<gblk(Nn), TPB, 0, stream>>>(deg, dis, Nn);
  edge_norms<<<gblk(Ecnt), TPB, 0, stream>>>(esrc, edst, ew, dis, enorm, Ecnt);

  // --- GCN layer 1 ---
  cvt(x, XB, (size_t)Nn * D);
  gemm(XB, wg1, G, Nn, D, Hh);
  fill_f32<<<gblk((size_t)Nn * Hh), TPB, 0, stream>>>(T, 0.0f, (size_t)Nn * Hh);
  { dim3 g(Ecnt, Hh / TPB);
    edge_aggregate<<<g, TPB, 0, stream>>>(G, T, esrc, edst, enorm, Hh); }
  agg_finish<<<gblk((size_t)Nn * Hh), TPB, 0, stream>>>(T, G, dis, gcn1_b, Nn, Hh);
  fill_f32<<<gblk(1024), TPB, 0, stream>>>(ssum, 0.0f, 1024);
  fill_f32<<<gblk(1024), TPB, 0, stream>>>(ssq,  0.0f, 1024);
  bn_colstats<<<(Nn + 127) / 128, 256, 0, stream>>>(T, ssum, ssq, Nn, 128);
  bn_normalize_bf16<<<gblk((size_t)Nn * Hh), TPB, 0, stream>>>(T, ssum, ssq, bn1_g, bn1_b, XB, Nn, Hh);

  // --- GCN layer 2 ---
  gemm(XB, wg2, G, Nn, Hh, Hh);
  fill_f32<<<gblk((size_t)Nn * Hh), TPB, 0, stream>>>(T, 0.0f, (size_t)Nn * Hh);
  { dim3 g(Ecnt, Hh / TPB);
    edge_aggregate<<<g, TPB, 0, stream>>>(G, T, esrc, edst, enorm, Hh); }
  agg_finish<<<gblk((size_t)Nn * Hh), TPB, 0, stream>>>(T, G, dis, gcn2_b, Nn, Hh);
  fill_f32<<<gblk(1024), TPB, 0, stream>>>(ssum, 0.0f, 1024);
  fill_f32<<<gblk(1024), TPB, 0, stream>>>(ssq,  0.0f, 1024);
  bn_colstats<<<(Nn + 127) / 128, 256, 0, stream>>>(T, ssum, ssq, Nn, 128);
  bn_normalize_bf16<<<gblk((size_t)Nn * Hh), TPB, 0, stream>>>(T, ssum, ssq, bn2_g, bn2_b, XB, Nn, Hh);

  // --- projection to D=512 ---
  gemm(XB, wproj, G, Nn, Hh, D);
  epilogue_f32_bf16<<<gblk((size_t)Nn * D), TPB, 0, stream>>>(G, proj_b, Hb, XB, Nn, D);

  // --- transformer layers ---
  for (int l = 0; l < Lc; ++l) {
    const float* bv = t_qkv_b + (size_t)l * 3 * D + 2 * D;
    gemm(XB, wv[l], G, Nn, D, D);
    epilogue_bf16<<<gblk((size_t)Nn * D), TPB, 0, stream>>>(G, bv, XB, Nn, D, 0);
    gemm(XB, wout[l], G, Nn, D, D);
    ln_fused<<<(Nn + 7) / 8, 256, 0, stream>>>(Hb, G, t_out_b + (size_t)l * D,
                                               t_ln1_g + (size_t)l * D, t_ln1_b + (size_t)l * D, XB, Nn);
    gemm(XB, wff1[l], G, Nn, D, DFF);
    epilogue_bf16<<<gblk((size_t)Nn * DFF), TPB, 0, stream>>>(G, t_ff1_b + (size_t)l * DFF, XB, Nn, DFF, 1);
    gemm(XB, wff2[l], G, Nn, DFF, D);
    ln_fused<<<(Nn + 7) / 8, 256, 0, stream>>>(Hb, G, t_ff2_b + (size_t)l * D,
                                               t_ln2_g + (size_t)l * D, t_ln2_b + (size_t)l * D, XB, Nn);
  }

  // --- head ---
  gemm(XB, wfc, G, Nn, D, D);
  epilogue_bf16<<<gblk((size_t)Nn * D), TPB, 0, stream>>>(G, fc_b, XB, Nn, D, 1);
  fc1_kernel<<<gblk(Nn), TPB, 0, stream>>>(XB, fc1_w, fc1_b, (float*)d_out, Nn);
}
